// SingleAttention_78099685310952
// MI455X (gfx1250) — matrix operand order repaired and checked by reading, not compiled, not run on hardware
//
#include <hip/hip_runtime.h>
#include <stdint.h>

typedef __bf16 v16bf __attribute__((ext_vector_type(16)));
typedef __bf16 v8bf  __attribute__((ext_vector_type(8)));
typedef float  v8f   __attribute__((ext_vector_type(8)));
typedef int    v4i   __attribute__((ext_vector_type(4)));

#define DEVINL static __device__ __forceinline__

union BF16x16 { v16bf v; v8bf h[2]; };

// ---------------------------------------------------------------------------
// CDNA5 async global->LDS path (ASYNCcnt), with safe fallback
// ---------------------------------------------------------------------------
#if __has_builtin(__builtin_amdgcn_global_load_async_to_lds_b128) && \
    __has_builtin(__builtin_amdgcn_s_wait_asynccnt)
#define HAS_ASYNC 1
#else
#define HAS_ASYNC 0
#endif

typedef __attribute__((address_space(1))) v4i gv4i_t;  // global (device) v4i
typedef __attribute__((address_space(3))) v4i lv4i_t;  // LDS v4i

DEVINL void copy16(const __bf16* g, __bf16* l) {
#if HAS_ASYNC
  // emits global_load_async_to_lds_b128 (ASYNCcnt-tracked DMA into LDS)
  __builtin_amdgcn_global_load_async_to_lds_b128(
      (gv4i_t*)(uintptr_t)g, (lv4i_t*)(uint32_t)(uintptr_t)l, 0, 0);
#else
  *(v8bf*)l = *(const v8bf*)g;
#endif
}

#if HAS_ASYNC
#define WAIT_ASYNC(n) __builtin_amdgcn_s_wait_asynccnt(n)
#else
#define WAIT_ASYNC(n) ((void)0)
#endif

// ---------------------------------------------------------------------------
// CDNA5 Tensor Data Mover path (TENSORcnt), with fallback to async staging
// ---------------------------------------------------------------------------
#if __has_builtin(__builtin_amdgcn_tensor_load_to_lds) && \
    __has_builtin(__builtin_amdgcn_s_wait_tensorcnt)
#define HAS_TDM 1
#else
#define HAS_TDM 0
#endif

#if HAS_TDM
typedef unsigned int u32x4 __attribute__((ext_vector_type(4)));
typedef int          i32x4 __attribute__((ext_vector_type(4)));
typedef int          i32x8 __attribute__((ext_vector_type(8)));

// One TDM descriptor: contiguous 8KB tile = 1-D tensor of 4096 x 2-byte elems.
DEVINL void tdm_load_tile(const __bf16* gsrc, __bf16* ldst) {
  const uint64_t ga  = (uint64_t)(uintptr_t)gsrc;     // 57-bit global byte addr
  const uint32_t lds = (uint32_t)(uintptr_t)ldst;     // LDS byte offset
  u32x4 g0;
  g0[0] = 1u;                                               // count=1, user mode
  g0[1] = lds;                                              // lds_addr  [63:32]
  g0[2] = (uint32_t)ga;                                     // global_addr[95:64]
  g0[3] = (uint32_t)((ga >> 32) & 0x01FFFFFFu) | (2u << 30);// ga[56:32], type=2
  i32x8 g1;
  g1[0] = 0x00010000;          // workgroup_mask=0, data_size=1 (2 bytes)
  g1[1] = (int)(4096u << 16);  // tensor_dim0[15:0] = 4096   (bits [79:48])
  g1[2] = (int)(1u << 16);     // tensor_dim0 hi = 0, tensor_dim1[15:0] = 1
  g1[3] = (int)(4096u << 16);  // tensor_dim1 hi = 0, tile_dim0 = 4096
  g1[4] = 0;                   // tile_dim1 = 0 (unused), tile_dim2 = 0
  g1[5] = 4096;                // tensor_dim0_stride[31:0]
  g1[6] = 0;
  g1[7] = 0;
  i32x4 gz = {0, 0, 0, 0};     // groups 2/3 unused (<=2D tensor)
#if __clang_major__ >= 23
  i32x8 gz8 = {};
  __builtin_amdgcn_tensor_load_to_lds(g0, g1, gz, gz, gz8, 0);
#else
  __builtin_amdgcn_tensor_load_to_lds(g0, g1, gz, gz, 0);
#endif
}
#define WAIT_TENSOR(n) __builtin_amdgcn_s_wait_tensorcnt(n)
#endif

DEVINL v8f wmma_bf16(v16bf a, v16bf b, v8f c) {
  // D = A(16x32) * B(32x16) + C(16x16), fp32 accumulate
  return __builtin_amdgcn_wmma_f32_16x16x32_bf16(
      /*neg_a=*/false, a, /*neg_b=*/false, b,
      /*c_mod=*/(short)0, c, /*reuse_a=*/false, /*reuse_b=*/false);
}

DEVINL v8bf cvt8(float4 a, float4 b) {
  v8bf r;
  r[0] = (__bf16)a.x; r[1] = (__bf16)a.y; r[2] = (__bf16)a.z; r[3] = (__bf16)a.w;
  r[4] = (__bf16)b.x; r[5] = (__bf16)b.y; r[6] = (__bf16)b.z; r[7] = (__bf16)b.w;
  return r;
}

constexpr int BATCH = 8;
constexpr int SEQ   = 2048;
constexpr int DIN   = 512;
constexpr int DKV   = 64;
constexpr int NROWS = BATCH * SEQ;  // 16384

// ---------------------------------------------------------------------------
// Kernel 0: W (fp32 [512][64]) -> Wt (bf16 [64][512]), for q/k/v (blockIdx.y)
// ---------------------------------------------------------------------------
__global__ void wtrans_kernel(const float* __restrict__ Wq,
                              const float* __restrict__ Wk,
                              const float* __restrict__ Wv,
                              __bf16* __restrict__ WtAll) {
  const int which = blockIdx.y;
  const float* W = (which == 0) ? Wq : (which == 1) ? Wk : Wv;
  __bf16* O = WtAll + (size_t)which * DKV * DIN;
  const int o = blockIdx.x * blockDim.x + threadIdx.x;  // 0..32767
  const int n = o >> 9;       // 0..63
  const int k = o & 511;      // 0..511
  O[o] = (__bf16)W[k * DKV + n];
}

// ---------------------------------------------------------------------------
// Kernel 1: projection  Out[which] = bf16( X[which] @ W[which] + b[which] )
// 256 threads = 8 waves; wave: 16 rows x 64 cols, K=512.
// W^T staged to LDS via async DMA; A-fragment stream double-buffered.
// ---------------------------------------------------------------------------
DEVINL BF16x16 load_a_frag(const float* xr, int kc, int kbA) {
  float4 f0 = *(const float4*)(xr + kc + kbA + 0);
  float4 f1 = *(const float4*)(xr + kc + kbA + 4);
  float4 f2 = *(const float4*)(xr + kc + kbA + 8);
  float4 f3 = *(const float4*)(xr + kc + kbA + 12);
  BF16x16 a;
  a.h[0] = cvt8(f0, f1);
  a.h[1] = cvt8(f2, f3);
  return a;
}

__global__ void __launch_bounds__(256) proj_kernel(
    const float* __restrict__ Xq, const float* __restrict__ Xk,
    const float* __restrict__ Xv, const __bf16* __restrict__ WtAll,
    const float* __restrict__ bq, const float* __restrict__ bk,
    const float* __restrict__ bv, __bf16* __restrict__ OutAll) {
  const int which = blockIdx.y;
  const float* X    = (which == 0) ? Xq : (which == 1) ? Xk : Xv;
  const float* bias = (which == 0) ? bq : (which == 1) ? bk : bv;
  const __bf16* Wt  = WtAll + (size_t)which * DKV * DIN;
  __bf16* O         = OutAll + (size_t)which * NROWS * DKV;

  __shared__ __attribute__((aligned(32))) __bf16 Wlds[DKV * DIN];  // 64KB

  // Stage W^T into LDS: 32768 elems / 256 threads / 8-per-op = 16 async ops/thread
#pragma unroll
  for (int it = 0; it < 16; ++it) {
    const int off = (threadIdx.x + it * 256) * 8;
    copy16(Wt + off, Wlds + off);
  }
  WAIT_ASYNC(0);
  __syncthreads();

  const int lane = threadIdx.x & 31;
  const int wave = threadIdx.x >> 5;
  const int col  = lane & 15;
  const bool hi  = lane >= 16;
  const int kbA  = hi ? 16 : 0;   // A-fragment lane K-base
  const int kbB  = hi ? 16 : 0;   // B-fragment lane K-base
  const int row_base = blockIdx.x * 128 + wave * 16;

  const float* xr = X + (size_t)(row_base + col) * DIN;

  v8f acc[4] = {};
  BF16x16 a_cur = load_a_frag(xr, 0, kbA);
#pragma unroll 2
  for (int kc = 0; kc < DIN; kc += 32) {
    BF16x16 a_nxt = a_cur;
    if (kc + 32 < DIN) a_nxt = load_a_frag(xr, kc + 32, kbA);  // prefetch next chunk
#pragma unroll
    for (int t = 0; t < 4; ++t) {
      v16bf bfrag = *(const v16bf*)(Wlds + (size_t)(t * 16 + col) * DIN + kc + kbB);
      acc[t] = wmma_bf16(a_cur.v, bfrag, acc[t]);
    }
    a_cur = a_nxt;
  }

#pragma unroll
  for (int r = 0; r < 8; ++r) {
    const int row = row_base + (hi ? 8 + r : r);
#pragma unroll
    for (int t = 0; t < 4; ++t) {
      O[(size_t)row * DKV + t * 16 + col] = (__bf16)(acc[t][r] + bias[t * 16 + col]);
    }
  }
}

// ---------------------------------------------------------------------------
// Kernel 2: flash attention. grid = (SEQ/64, BATCH), block = 128 (4 waves).
// K/V tiles double-buffered in LDS; wave 0 drives the Tensor Data Mover (or
// all waves use async DMA as fallback). Next tile streams during compute.
// ---------------------------------------------------------------------------
#if !HAS_TDM
DEVINL void stage_tile(const __bf16* Ksrc, const __bf16* Vsrc,
                       __bf16* Kdst, __bf16* Vdst, int tid) {
  // 64x64 bf16 tile is contiguous (8KB); 128 threads x 4 x 16B each
#pragma unroll
  for (int it = 0; it < 4; ++it) {
    const int off = (tid + it * 128) * 8;
    copy16(Ksrc + off, Kdst + off);
  }
#pragma unroll
  for (int it = 0; it < 4; ++it) {
    const int off = (tid + it * 128) * 8;
    copy16(Vsrc + off, Vdst + off);
  }
}
#endif

__global__ void __launch_bounds__(128) attn_kernel(
    const __bf16* __restrict__ Qb, const __bf16* __restrict__ Kb,
    const __bf16* __restrict__ Vb, float* __restrict__ Out) {
  constexpr float scale = 0.125f;  // 1/sqrt(64)
  const int tid  = threadIdx.x;
  const int lane = tid & 31;
  const int wave = tid >> 5;
  const int col  = lane & 15;
  const bool hi  = lane >= 16;
  const int kbA  = hi ? 16 : 0;
  const int kbB  = hi ? 16 : 0;
  const int batch = blockIdx.y;
  const int q0    = blockIdx.x * 64 + wave * 16;

  const __bf16* Q  = Qb + ((size_t)batch * SEQ + q0) * DKV;
  const __bf16* Km = Kb + (size_t)batch * SEQ * DKV;
  const __bf16* Vm = Vb + (size_t)batch * SEQ * DKV;

  __shared__ __attribute__((aligned(32))) __bf16 Kt[2][64 * DKV];    // [buf][key][dim]
  __shared__ __attribute__((aligned(32))) __bf16 Vrow[2][64 * DKV];  // [buf][key][dim]
  __shared__ __attribute__((aligned(32))) __bf16 Vt[DKV * 64];       // [dim][key]
  __shared__ __attribute__((aligned(32))) __bf16 Pl[4][16 * 64];     // per-wave P

  // Q A-fragments (K=64 -> 2 chunks of 32), loaded once
  BF16x16 aq[2];
  {
    const __bf16* qr = Q + (size_t)col * DKV;
#pragma unroll
    for (int c = 0; c < 2; ++c) {
      aq[c].h[0] = *(const v8bf*)(qr + c * 32 + kbA + 0);
      aq[c].h[1] = *(const v8bf*)(qr + c * 32 + kbA + 8);
    }
  }

  v8f o[4] = {};
  float m_r[8], l_r[8];
#pragma unroll
  for (int r = 0; r < 8; ++r) { m_r[r] = -3.0e38f; l_r[r] = 0.0f; }

  // Kick off tile 0
#if HAS_TDM
  if (wave == 0) {
    tdm_load_tile(Km, &Kt[0][0]);
    tdm_load_tile(Vm, &Vrow[0][0]);
  }
#else
  stage_tile(Km, Vm, Kt[0], Vrow[0], tid);
#endif

  for (int j = 0; j < SEQ; j += 64) {
    const int cur = (j >> 6) & 1;
    const int nxt = cur ^ 1;

    __syncthreads();  // all waves done with buffers [nxt] + Vt from prev iter
#if HAS_TDM
    if (wave == 0) {
      if (j + 64 < SEQ) {
        tdm_load_tile(Km + (size_t)(j + 64) * DKV, &Kt[nxt][0]);
        tdm_load_tile(Vm + (size_t)(j + 64) * DKV, &Vrow[nxt][0]);
        WAIT_TENSOR(2);  // tensor ops in-order: tile [cur] is complete
      } else {
        WAIT_TENSOR(0);
      }
    }
#else
    if (j + 64 < SEQ) {
      stage_tile(Km + (size_t)(j + 64) * DKV, Vm + (size_t)(j + 64) * DKV,
                 Kt[nxt], Vrow[nxt], tid);
      WAIT_ASYNC(8);  // only the 8 just-issued remain -> tile [cur] is complete
    } else {
      WAIT_ASYNC(0);
    }
#endif
    __syncthreads();

    // LDS->LDS transpose: Vt[dim][key] = Vrow[cur][key][dim]
#pragma unroll
    for (int it = 0; it < 4; ++it) {
      const int w   = tid + it * 128;  // 0..511
      const int key = w >> 3;
      const int g   = w & 7;
      v8bf vv = *(const v8bf*)(&Vrow[cur][key * DKV + g * 8]);
#pragma unroll
      for (int d = 0; d < 8; ++d) Vt[(g * 8 + d) * 64 + key] = vv[d];
    }
    __syncthreads();

    // scores = q @ k^T : B-fragment of k^T == contiguous K-tile rows in LDS
    const __bf16* Kc = Kt[cur];
    v8f s[4] = {};
#pragma unroll
    for (int t = 0; t < 4; ++t) {
      const __bf16* kr = Kc + (size_t)(t * 16 + col) * DKV;
#pragma unroll
      for (int c = 0; c < 2; ++c) {
        v16bf bkf = *(const v16bf*)(kr + c * 32 + kbB);
        s[t] = wmma_bf16(aq[c].v, bkf, s[t]);
      }
    }

    // online softmax (row stats via half-wave xor reductions), P -> LDS bf16
    __bf16* Pw = Pl[wave];
#pragma unroll
    for (int r = 0; r < 8; ++r) {
      float v0 = s[0][r] * scale, v1 = s[1][r] * scale;
      float v2 = s[2][r] * scale, v3 = s[3][r] * scale;
      float mx = fmaxf(fmaxf(v0, v1), fmaxf(v2, v3));
      mx = fmaxf(mx, __shfl_xor(mx, 1));
      mx = fmaxf(mx, __shfl_xor(mx, 2));
      mx = fmaxf(mx, __shfl_xor(mx, 4));
      mx = fmaxf(mx, __shfl_xor(mx, 8));
      const float mn   = fmaxf(m_r[r], mx);
      const float corr = __expf(m_r[r] - mn);
      const float p0 = __expf(v0 - mn), p1 = __expf(v1 - mn);
      const float p2 = __expf(v2 - mn), p3 = __expf(v3 - mn);
      float rs = p0 + p1 + p2 + p3;
      rs += __shfl_xor(rs, 1);
      rs += __shfl_xor(rs, 2);
      rs += __shfl_xor(rs, 4);
      rs += __shfl_xor(rs, 8);
      l_r[r] = l_r[r] * corr + rs;
      m_r[r] = mn;
#pragma unroll
      for (int t = 0; t < 4; ++t) o[t][r] *= corr;
      const int row = hi ? 8 + r : r;
      Pw[row * 64 +  0 + col] = (__bf16)p0;
      Pw[row * 64 + 16 + col] = (__bf16)p1;
      Pw[row * 64 + 32 + col] = (__bf16)p2;
      Pw[row * 64 + 48 + col] = (__bf16)p3;
    }

    __builtin_amdgcn_wave_barrier();  // wave-private LDS: order P stores vs loads

    // o += P(16x64) @ Vtile(64x64)
    BF16x16 ap[2];
    {
      const __bf16* pr = Pw + (size_t)col * 64;
#pragma unroll
      for (int c = 0; c < 2; ++c) {
        ap[c].h[0] = *(const v8bf*)(pr + c * 32 + kbA + 0);
        ap[c].h[1] = *(const v8bf*)(pr + c * 32 + kbA + 8);
      }
    }
#pragma unroll
    for (int t = 0; t < 4; ++t) {
      const __bf16* vr = Vt + (size_t)(t * 16 + col) * 64;
#pragma unroll
      for (int c = 0; c < 2; ++c) {
        v16bf bvf = *(const v16bf*)(vr + c * 32 + kbB);
        o[t] = wmma_bf16(ap[c].v, bvf, o[t]);
      }
    }
  }

  float* Ob = Out + ((size_t)batch * SEQ + q0) * DKV;
#pragma unroll
  for (int r = 0; r < 8; ++r) {
    const int row = hi ? 8 + r : r;
    const float inv = 1.0f / l_r[r];
#pragma unroll
    for (int t = 0; t < 4; ++t) {
      Ob[(size_t)row * DKV + t * 16 + col] = o[t][r] * inv;
    }
  }
}

// ---------------------------------------------------------------------------
extern "C" void kernel_launch(void* const* d_in, const int* in_sizes, int n_in,
                              void* d_out, int out_size, void* d_ws, size_t ws_size,
                              hipStream_t stream) {
  const float* q_in = (const float*)d_in[0];
  const float* k_in = (const float*)d_in[1];
  const float* v_in = (const float*)d_in[2];
  const float* Wq   = (const float*)d_in[3];
  const float* bq   = (const float*)d_in[4];
  const float* Wk   = (const float*)d_in[5];
  const float* bk   = (const float*)d_in[6];
  const float* Wv   = (const float*)d_in[7];
  const float* bv   = (const float*)d_in[8];

  // workspace (bf16 elems): Wq^T|Wk^T|Wv^T then Q|K|V
  __bf16* ws16 = (__bf16*)d_ws;
  __bf16* Wt   = ws16;
  __bf16* Qb   = ws16 + (size_t)3 * DKV * DIN;
  __bf16* Kb   = Qb + (size_t)NROWS * DKV;
  __bf16* Vb   = Kb + (size_t)NROWS * DKV;

  wtrans_kernel<<<dim3(128, 3), 256, 0, stream>>>(Wq, Wk, Wv, Wt);
  proj_kernel<<<dim3(128, 3), 256, 0, stream>>>(q_in, k_in, v_in, Wt, bq, bk, bv, Qb);
  attn_kernel<<<dim3(SEQ / 64, BATCH), 128, 0, stream>>>(Qb, Kb, Vb, (float*)d_out);
}
